// GNNClassifier_88648124990098
// MI455X (gfx1250) — compile-verified, matrix operand order checked by reading, not run on hardware
//
#include <hip/hip_runtime.h>
#include <hip/hip_bf16.h>

// ---------------------------------------------------------------------------
// GNN SAGEConv x2 + pool + linear for MI455X (gfx1250, wave32, WMMA)
// ---------------------------------------------------------------------------

#define N_NODES   50000
#define N_EDGES   800000
#define EMBED     128
#define HIDDEN    128
#define N_CLASSES 10
#define N_GRAPHS  64
#define KDIM      256   // concat(agg, x)

typedef __attribute__((ext_vector_type(16))) _Float16 v16h;
typedef __attribute__((ext_vector_type(8)))  _Float16 v8h;
typedef __attribute__((ext_vector_type(8)))  float    v8f;

// ---------------------------------------------------------------------------
// Elementwise helpers
// ---------------------------------------------------------------------------

__global__ __launch_bounds__(256) void zero_f_kernel(float* __restrict__ p, int n) {
    int i = blockIdx.x * 256 + threadIdx.x;
    if (i < n) p[i] = 0.0f;
}

// Embedding gather: X[n][:] = table[x_idx[n]][:]   (one wave per node, float4/lane)
__global__ __launch_bounds__(256) void embed_kernel(const int* __restrict__ idx,
                                                    const float* __restrict__ table,
                                                    float* __restrict__ X) {
    int lane = threadIdx.x & 31;
    int n = blockIdx.x * 8 + (threadIdx.x >> 5);
    if (n >= N_NODES) return;
    int row = idx[n];
    float4 v = *(const float4*)(table + (size_t)row * EMBED + lane * 4);
    *(float4*)(X + (size_t)n * EMBED + lane * 4) = v;
}

// Edge scatter: msg[dst][:] += X[src][:], cnt[dst] += 1  (one wave per edge)
__global__ __launch_bounds__(256) void scatter_kernel(const int* __restrict__ ei,
                                                      const float* __restrict__ X,
                                                      float* __restrict__ msg,
                                                      float* __restrict__ cnt) {
    int lane   = threadIdx.x & 31;
    int warp   = blockIdx.x * 8 + (threadIdx.x >> 5);
    int nWarps = gridDim.x * 8;
    for (int e = warp; e < N_EDGES; e += nWarps) {
        int s = ei[e];
        int d = ei[N_EDGES + e];
        float4 v = *(const float4*)(X + (size_t)s * HIDDEN + lane * 4);
        float* mp = msg + (size_t)d * HIDDEN + lane * 4;
        atomicAdd(mp + 0, v.x);
        atomicAdd(mp + 1, v.y);
        atomicAdd(mp + 2, v.z);
        atomicAdd(mp + 3, v.w);
        if (lane == 0) atomicAdd(cnt + d, 1.0f);
    }
}

// Build fp16 activation matrix: act[n][0:128] = msg[n]/max(cnt,1), act[n][128:256] = X[n]
__global__ __launch_bounds__(256) void build_act_kernel(const float* __restrict__ msg,
                                                        const float* __restrict__ cnt,
                                                        const float* __restrict__ X,
                                                        _Float16* __restrict__ act) {
    int n = blockIdx.x;
    int k = threadIdx.x;           // 0..255
    float c = cnt[n];
    c = (c > 1.0f) ? c : 1.0f;
    float v;
    if (k < HIDDEN) v = msg[(size_t)n * HIDDEN + k] / c;
    else            v = X[(size_t)n * HIDDEN + (k - HIDDEN)];
    act[(size_t)n * KDIM + k] = (_Float16)v;
}

// Build fp16 combined weight: wcomb[h][0:128] = Wl[h], wcomb[h][128:256] = Wr[h]
__global__ __launch_bounds__(256) void build_w_kernel(const float* __restrict__ Wl,
                                                      const float* __restrict__ Wr,
                                                      _Float16* __restrict__ wcomb) {
    int i = blockIdx.x * 256 + threadIdx.x;   // 128*256 total
    int h = i >> 8;
    int k = i & 255;
    float v = (k < HIDDEN) ? Wl[h * HIDDEN + k] : Wr[h * HIDDEN + (k - HIDDEN)];
    wcomb[i] = (_Float16)v;
}

// ---------------------------------------------------------------------------
// WMMA GEMM: H = relu(ACT[N,256] @ Wcomb[128,256]^T + bias)
//   Block: 256 threads (8 waves). grid = (ceil(N/128), 2).
//   Each block stages a 64x256 fp16 weight tile in LDS (padded, conflict-free),
//   each wave owns a 16-row strip x 64 cols (4 accumulators of 16x16).
// ---------------------------------------------------------------------------
#define LDW 264   // padded LDS row stride (halfs): 528B = 33*16B -> no bank conflicts

__global__ __launch_bounds__(256)
void sage_gemm_kernel(const _Float16* __restrict__ act,     // [N, 256]
                      const _Float16* __restrict__ wcomb,   // [128, 256]
                      const float*    __restrict__ bias,    // [128]
                      float*          __restrict__ out) {   // [N, 128]
    __shared__ __align__(16) _Float16 wlds[64 * LDW];

    const int tid      = threadIdx.x;
    const int colBlock = blockIdx.y;          // cols [64*colBlock, 64*colBlock+64)

    // Stage 64x256 weight tile into LDS (dword granularity, fully coalesced).
    for (int i = tid; i < 64 * 128; i += 256) {
        int r = i >> 7;        // weight row within tile (output channel)
        int c = i & 127;       // dword within row (2 halfs)
        const uint32_t* src = (const uint32_t*)(wcomb + (size_t)(colBlock * 64 + r) * KDIM) + c;
        *(uint32_t*)(&wlds[r * LDW + c * 2]) = *src;
    }
    __syncthreads();

    const int wave    = tid >> 5;
    const int lane    = tid & 31;
    const int rowBase = blockIdx.x * 128 + wave * 16;
    if (rowBase >= N_NODES) return;           // wave-uniform: EXEC stays all-ones

    const int  lrow = lane & 15;
    const bool hi   = lane >= 16;
    // ISA A-layout (16-bit A, 16x32): lanes 0-15 hold K{0..7,16..23}, lanes 16-31 K{8..15,24..31}
    const int kOff0 = hi ? 8 : 0;
    const int kOff1 = hi ? 24 : 16;

    const _Float16* arow = act + (size_t)(rowBase + lrow) * KDIM;

    v8f acc[4] = {};

    #pragma unroll
    for (int kt = 0; kt < 8; ++kt) {
        const int k0 = kt * 32;
        v8h a0 = *(const v8h*)(arow + k0 + kOff0);
        v8h a1 = *(const v8h*)(arow + k0 + kOff1);
        v16h a = __builtin_shufflevector(a0, a1, 0, 1, 2, 3, 4, 5, 6, 7,
                                                 8, 9, 10, 11, 12, 13, 14, 15);
        #pragma unroll
        for (int ct = 0; ct < 4; ++ct) {
            const _Float16* brow = &wlds[(ct * 16 + lrow) * LDW + k0];
            v8h b0 = *(const v8h*)(brow + kOff0);
            v8h b1 = *(const v8h*)(brow + kOff1);
            v16h b = __builtin_shufflevector(b0, b1, 0, 1, 2, 3, 4, 5, 6, 7,
                                                     8, 9, 10, 11, 12, 13, 14, 15);
            // D = A x B + C, fp32 accumulate
            acc[ct] = __builtin_amdgcn_wmma_f32_16x16x32_f16(
                false, a, false, b, (short)0, acc[ct], false, false);
        }
    }

    // Epilogue: bias + ReLU. C/D layout: VGPR j -> M = j (+8 for lanes>=16), lane&15 -> N.
    #pragma unroll
    for (int ct = 0; ct < 4; ++ct) {
        int col = colBlock * 64 + ct * 16 + lrow;
        float bv = bias[col];
        #pragma unroll
        for (int j = 0; j < 8; ++j) {
            int row = rowBase + j + (hi ? 8 : 0);
            float v = acc[ct][j] + bv;
            out[(size_t)row * HIDDEN + col] = v > 0.0f ? v : 0.0f;
        }
    }
}

// ---------------------------------------------------------------------------
// Global mean pool (scatter) + final linear
// ---------------------------------------------------------------------------
__global__ __launch_bounds__(256) void pool_kernel(const float* __restrict__ H,
                                                   const int* __restrict__ batch,
                                                   float* __restrict__ pooled,
                                                   float* __restrict__ gcnt) {
    int lane = threadIdx.x & 31;
    int n = blockIdx.x * 8 + (threadIdx.x >> 5);
    if (n >= N_NODES) return;
    int g = batch[n];
    float4 v = *(const float4*)(H + (size_t)n * HIDDEN + lane * 4);
    float* pp = pooled + (size_t)g * HIDDEN + lane * 4;
    atomicAdd(pp + 0, v.x);
    atomicAdd(pp + 1, v.y);
    atomicAdd(pp + 2, v.z);
    atomicAdd(pp + 3, v.w);
    if (lane == 0) atomicAdd(gcnt + g, 1.0f);
}

__global__ __launch_bounds__(256) void final_kernel(const float* __restrict__ pooled,
                                                    const float* __restrict__ gcnt,
                                                    const float* __restrict__ linW,
                                                    const float* __restrict__ linb,
                                                    float* __restrict__ out) {
    int t = blockIdx.x * 256 + threadIdx.x;
    if (t >= N_GRAPHS * N_CLASSES) return;
    int g = t / N_CLASSES;
    int c = t % N_CLASSES;
    float cg = gcnt[g];
    float inv = 1.0f / (cg > 1.0f ? cg : 1.0f);
    float s = 0.0f;
    #pragma unroll 4
    for (int d = 0; d < HIDDEN; ++d)
        s += pooled[g * HIDDEN + d] * linW[c * HIDDEN + d];
    out[t] = s * inv + linb[c];
}

// ---------------------------------------------------------------------------
// Launch orchestration
// ---------------------------------------------------------------------------
extern "C" void kernel_launch(void* const* d_in, const int* in_sizes, int n_in,
                              void* d_out, int out_size, void* d_ws, size_t ws_size,
                              hipStream_t stream) {
    const int*   x_idx      = (const int*)d_in[0];
    const int*   edge_index = (const int*)d_in[1];
    const int*   batch      = (const int*)d_in[2];
    const float* table      = (const float*)d_in[3];
    const float* Wl1        = (const float*)d_in[4];
    const float* bl1        = (const float*)d_in[5];
    const float* Wr1        = (const float*)d_in[6];
    const float* Wl2        = (const float*)d_in[7];
    const float* bl2        = (const float*)d_in[8];
    const float* Wr2        = (const float*)d_in[9];
    const float* linW       = (const float*)d_in[10];
    const float* linb       = (const float*)d_in[11];
    float* out = (float*)d_out;

    // Workspace carve-out (256B aligned)
    char* w = (char*)d_ws;
    auto carve = [&](size_t bytes) -> void* {
        void* p = (void*)w;
        w += (bytes + 255) & ~(size_t)255;
        return p;
    };
    const size_t featB = (size_t)N_NODES * HIDDEN * sizeof(float);   // 25.6 MB
    float*    bufA   = (float*)carve(featB);                         // x0 / h2
    float*    bufB   = (float*)carve(featB);                         // h1
    float*    msg    = (float*)carve(featB);
    _Float16* act    = (_Float16*)carve((size_t)N_NODES * KDIM * sizeof(_Float16));
    float*    cnt    = (float*)carve((size_t)N_NODES * sizeof(float));
    _Float16* wcomb  = (_Float16*)carve((size_t)HIDDEN * KDIM * sizeof(_Float16));
    float*    pooled = (float*)carve((size_t)(N_GRAPHS * HIDDEN + N_GRAPHS) * sizeof(float));
    float*    gcnt   = pooled + N_GRAPHS * HIDDEN;

    const int msgN   = N_NODES * HIDDEN;
    dim3 gemmGrid((N_NODES + 127) / 128, 2);

    // --- Embedding gather ---
    embed_kernel<<<(N_NODES + 7) / 8, 256, 0, stream>>>(x_idx, table, bufA);

    // --- SAGE layer 1 ---
    zero_f_kernel<<<(msgN + 255) / 256, 256, 0, stream>>>(msg, msgN);
    zero_f_kernel<<<(N_NODES + 255) / 256, 256, 0, stream>>>(cnt, N_NODES);
    scatter_kernel<<<(N_EDGES + 7) / 8, 256, 0, stream>>>(edge_index, bufA, msg, cnt);
    build_act_kernel<<<N_NODES, 256, 0, stream>>>(msg, cnt, bufA, act);
    build_w_kernel<<<(HIDDEN * KDIM) / 256, 256, 0, stream>>>(Wl1, Wr1, wcomb);
    sage_gemm_kernel<<<gemmGrid, 256, 0, stream>>>(act, wcomb, bl1, bufB);

    // --- SAGE layer 2 ---
    zero_f_kernel<<<(msgN + 255) / 256, 256, 0, stream>>>(msg, msgN);
    zero_f_kernel<<<(N_NODES + 255) / 256, 256, 0, stream>>>(cnt, N_NODES);
    scatter_kernel<<<(N_EDGES + 7) / 8, 256, 0, stream>>>(edge_index, bufB, msg, cnt);
    build_act_kernel<<<N_NODES, 256, 0, stream>>>(msg, cnt, bufB, act);
    build_w_kernel<<<(HIDDEN * KDIM) / 256, 256, 0, stream>>>(Wl2, Wr2, wcomb);
    sage_gemm_kernel<<<gemmGrid, 256, 0, stream>>>(act, wcomb, bl2, bufA);

    // --- Global mean pool + classifier ---
    zero_f_kernel<<<(N_GRAPHS * HIDDEN + N_GRAPHS + 255) / 256, 256, 0, stream>>>(
        pooled, N_GRAPHS * HIDDEN + N_GRAPHS);
    pool_kernel<<<(N_NODES + 7) / 8, 256, 0, stream>>>(bufA, batch, pooled, gcnt);
    final_kernel<<<(N_GRAPHS * N_CLASSES + 255) / 256, 256, 0, stream>>>(
        pooled, gcnt, linW, linb, out);
}
